// Scalarizer_52862457479883
// MI455X (gfx1250) — matrix operand order repaired and checked by reading, not compiled, not run on hardware
//
#include <hip/hip_runtime.h>
#include <stdint.h>

// ---------------------------------------------------------------------------
// Types for CDNA5 WMMA bf16 path
// ---------------------------------------------------------------------------
typedef __attribute__((ext_vector_type(16))) __bf16 bf16x16;
typedef __attribute__((ext_vector_type(8)))  float  f32x8;

union Frag {
  bf16x16 v;
  uint4   q[2];
};

__device__ __forceinline__ unsigned short f2bf(float x) {
  uint32_t u = __float_as_uint(x);
  u += 0x7fffu + ((u >> 16) & 1u);           // round-to-nearest-even to bf16
  return (unsigned short)(u >> 16);
}
__device__ __forceinline__ float bf2f(unsigned short h) {
  return __uint_as_float(((uint32_t)h) << 16);
}
__device__ __forceinline__ uint32_t pack2(unsigned short a, unsigned short b) {
  return (uint32_t)a | ((uint32_t)b << 16);
}
__device__ __forceinline__ void storeSplit(unsigned short* hi, unsigned short* lo,
                                           int idx, float x) {
  unsigned short h = f2bf(x);
  hi[idx] = h;
  lo[idx] = f2bf(x - bf2f(h));
}
__device__ __forceinline__ void loadFragLDS(Frag& f, const unsigned short* base) {
  f.q[0] = *(const uint4*)(base);        // K = kb .. kb+7   (2 bf16 / dword)
  f.q[1] = *(const uint4*)(base + 8);    // K = kb+8 .. kb+15
}
__device__ __forceinline__ void loadFragG(Frag& f, const uint32_t* p) {
  const uint4* q = (const uint4*)p;
  f.q[0] = q[0];
  f.q[1] = q[1];
}
__device__ __forceinline__ f32x8 wmma_bf16(const Frag& a, const Frag& b, f32x8 c) {
  return __builtin_amdgcn_wmma_f32_16x16x32_bf16(false, a.v, false, b.v,
                                                 (short)0, c, false, false);
}

// ---------------------------------------------------------------------------
// Problem constants
// ---------------------------------------------------------------------------
#define NROWS   131072
#define HIDDEN  512
#define RED     128
#define OUT     512
#define EPS_N   1e-3f
#define LN_EPS  1e-5f

// Workspace layout (dwords). B matrices pre-swizzled into WMMA B-fragment
// layout: index = ((nt*16 + kt)*32 + lane)*8 + d ; lane: N=lane&15, K-half=lane>>4,
// dword d holds K = kt*32 + (lane>>4)*16 + 2d (+1 in high half).
#define OFF_B1HI 0u        // W_dimred^T hi : 8 ntiles * 16 ktiles -> 32768 dw
#define OFF_B1LO 32768u
#define OFF_B2HI 65536u    // W_lin^T    hi : 32 ntiles * 16 ktiles -> 131072 dw
#define OFF_B2LO 196608u

#define SMEM_BYTES 265216  // 133120 (v2 hi/lo) + 132096 (bufA x2 | vred | h)

// ---------------------------------------------------------------------------
// Prep kernel: split weights into bf16 hi/lo, pre-swizzled B fragments
// ---------------------------------------------------------------------------
__global__ void prep_weights(const float* __restrict__ Wd,
                             const float* __restrict__ Wl,
                             uint32_t* __restrict__ ws) {
  int idx = blockIdx.x * 256 + threadIdx.x;
  if (idx < 32768) {                       // B1 = W_dimred^T : K=512, N=128
    int d = idx & 7, lane = (idx >> 3) & 31, t = idx >> 8;
    int kt = t & 15, nt = t >> 4;
    int r  = nt * 16 + (lane & 15);
    int k0 = kt * 32 + (lane >> 4) * 16 + d * 2;
    float f0 = Wd[r * HIDDEN + k0];
    float f1 = Wd[r * HIDDEN + k0 + 1];
    unsigned short h0 = f2bf(f0), h1 = f2bf(f1);
    ws[OFF_B1HI + idx] = pack2(h0, h1);
    ws[OFF_B1LO + idx] = pack2(f2bf(f0 - bf2f(h0)), f2bf(f1 - bf2f(h1)));
  } else if (idx < 163840) {               // B2 = W_lin^T : K=512, N=512
    int j = idx - 32768;
    int d = j & 7, lane = (j >> 3) & 31, t = j >> 8;
    int kt = t & 15, nt = t >> 4;
    int o  = nt * 16 + (lane & 15);
    int k0 = kt * 32 + (lane >> 4) * 16 + d * 2;
    float f0 = Wl[o * 512 + k0];
    float f1 = Wl[o * 512 + k0 + 1];
    unsigned short h0 = f2bf(f0), h1 = f2bf(f1);
    ws[OFF_B2HI + j] = pack2(h0, h1);
    ws[OFF_B2LO + j] = pack2(f2bf(f0 - bf2f(h0)), f2bf(f1 - bf2f(h1)));
  }
}

// ---------------------------------------------------------------------------
// Fused main kernel: 64 rows / workgroup, 256 threads (8 wave32)
// ---------------------------------------------------------------------------
__global__ __launch_bounds__(256, 1)
void scalarizer_fused(const float* __restrict__ v,
                      const float* __restrict__ W_emb,
                      const float* __restrict__ b_lin,
                      const float* __restrict__ gamma,
                      const float* __restrict__ beta,
                      const uint32_t* __restrict__ ws,
                      float* __restrict__ out) {
  extern __shared__ char smem[];
  unsigned short* v2hi = (unsigned short*)smem;         // [64][520] bf16
  unsigned short* v2lo = v2hi + 64 * 520;
  char*           R    = smem + 2 * 64 * 520 * 2;       // time-shared region
  // double-buffered split-A for GEMM1: buf{0,1}{hi,lo}, each [192][72] bf16
  unsigned short* bufS = (unsigned short*)R;            // 2*27648 ushorts
  float*          vred = (float*)R;                     // [192][132] f32
  float*          hbuf = (float*)R;                     // [64][516]  f32

  const int tid   = threadIdx.x;
  const int w     = tid >> 5;     // wave id 0..7
  const int lane  = tid & 31;
  const int m16   = lane & 15;
  const int khalf = lane >> 4;

  const float* vbase = v + (size_t)blockIdx.x * (64 * 3 * HIDDEN);
  const f32x8 zero8 = {0.f, 0.f, 0.f, 0.f, 0.f, 0.f, 0.f, 0.f};

  // ============== Phase A: GEMM1  (192x512)·(512x128), split-bf16 ==========
  // Software pipeline: register-stage next fp32 chunk while WMMAs consume the
  // current LDS buffer; convert+store into the other buffer after compute.
  f32x8 acc1[12];
#pragma unroll
  for (int t = 0; t < 12; ++t) acc1[t] = zero8;

  float4 vr[12];
  // prologue: chunk 0 -> buffer 0
#pragma unroll
  for (int it = 0; it < 12; ++it) {
    int idx = tid + it * 256;
    vr[it] = *(const float4*)(vbase + (idx >> 4) * HIDDEN + (idx & 15) * 4);
  }
  {
    unsigned short* hi0 = bufS;
    unsigned short* lo0 = bufS + 13824;
#pragma unroll
    for (int it = 0; it < 12; ++it) {
      int idx = tid + it * 256;
      int row = idx >> 4, c4 = idx & 15;
      float4 f = vr[it];
      unsigned short h0 = f2bf(f.x), h1 = f2bf(f.y), h2 = f2bf(f.z), h3 = f2bf(f.w);
      unsigned short l0 = f2bf(f.x - bf2f(h0)), l1 = f2bf(f.y - bf2f(h1));
      unsigned short l2 = f2bf(f.z - bf2f(h2)), l3 = f2bf(f.w - bf2f(h3));
      uint2 hv; hv.x = pack2(h0, h1); hv.y = pack2(h2, h3);
      uint2 lv; lv.x = pack2(l0, l1); lv.y = pack2(l2, l3);
      *(uint2*)(hi0 + row * 72 + c4 * 4) = hv;
      *(uint2*)(lo0 + row * 72 + c4 * 4) = lv;
    }
  }
  __syncthreads();

  for (int kc = 0; kc < 8; ++kc) {
    int cur = kc & 1;
    unsigned short* curHi = bufS + cur * 27648;
    unsigned short* curLo = curHi + 13824;
    // issue next chunk's global loads early (no wait until conversion)
    if (kc < 7) {
#pragma unroll
      for (int it = 0; it < 12; ++it) {
        int idx = tid + it * 256;
        vr[it] = *(const float4*)(vbase + (idx >> 4) * HIDDEN + (kc + 1) * 64 +
                                  (idx & 15) * 4);
      }
    }
    // compute from current buffer
#pragma unroll
    for (int ktl = 0; ktl < 2; ++ktl) {   // two K=32 tiles per chunk
      int ktg = kc * 2 + ktl;
      if (ktg < 15) {                      // prefetch next B1 fragment
        __builtin_prefetch(ws + OFF_B1HI + ((w * 16 + ktg + 1) * 32 + lane) * 8, 0, 3);
        __builtin_prefetch(ws + OFF_B1LO + ((w * 16 + ktg + 1) * 32 + lane) * 8, 0, 3);
      }
      Frag bh, bl;                         // wave w owns n-tile w of v_red
      loadFragG(bh, ws + OFF_B1HI + ((w * 16 + ktg) * 32 + lane) * 8);
      loadFragG(bl, ws + OFF_B1LO + ((w * 16 + ktg) * 32 + lane) * 8);
      int kb = ktl * 32 + khalf * 16;
#pragma unroll
      for (int mt = 0; mt < 12; ++mt) {
        Frag ah, al;
        int row = mt * 16 + m16;
        loadFragLDS(ah, curHi + row * 72 + kb);
        loadFragLDS(al, curLo + row * 72 + kb);
        f32x8 c = acc1[mt];
        c = wmma_bf16(ah, bh, c);          // hi*hi
        c = wmma_bf16(ah, bl, c);          // hi*lo
        c = wmma_bf16(al, bh, c);          // lo*hi
        acc1[mt] = c;
      }
    }
    // convert + store next chunk into the other buffer
    if (kc < 7) {
      unsigned short* nxtHi = bufS + (1 - cur) * 27648;
      unsigned short* nxtLo = nxtHi + 13824;
#pragma unroll
      for (int it = 0; it < 12; ++it) {
        int idx = tid + it * 256;
        int row = idx >> 4, c4 = idx & 15;
        float4 f = vr[it];
        unsigned short h0 = f2bf(f.x), h1 = f2bf(f.y), h2 = f2bf(f.z), h3 = f2bf(f.w);
        unsigned short l0 = f2bf(f.x - bf2f(h0)), l1 = f2bf(f.y - bf2f(h1));
        unsigned short l2 = f2bf(f.z - bf2f(h2)), l3 = f2bf(f.w - bf2f(h3));
        uint2 hv; hv.x = pack2(h0, h1); hv.y = pack2(h2, h3);
        uint2 lv; lv.x = pack2(l0, l1); lv.y = pack2(l2, l3);
        *(uint2*)(nxtHi + row * 72 + c4 * 4) = hv;
        *(uint2*)(nxtLo + row * 72 + c4 * 4) = lv;
      }
    }
    __syncthreads();
  }

  // dump v_red (192 x 128) f32 into LDS (row stride 132)
#pragma unroll
  for (int mt = 0; mt < 12; ++mt) {
#pragma unroll
    for (int j = 0; j < 8; ++j) {
      int m = mt * 16 + j + 8 * khalf;     // C layout: M = j + 8*(lane>>4)
      vred[m * 132 + w * 16 + m16] = acc1[mt][j];
    }
  }
  __syncthreads();

  // ============== Phase A2: norms / directions / projections -> v2 =========
  {
    int n = tid >> 2;                      // row 0..63, 4 lanes per row
    int q = tid & 3;                       // r-range q*32 .. q*32+31
    float dir[3][3] = {{0,0,0},{0,0,0},{0,0,0}};
    for (int rr = 0; rr < 32; ++rr) {
      int r = q * 32 + rr;
      float a0 = vred[(3 * n + 0) * 132 + r];
      float a1 = vred[(3 * n + 1) * 132 + r];
      float a2 = vred[(3 * n + 2) * 132 + r];
      float e0 = W_emb[0 * RED + r];
      float e1 = W_emb[1 * RED + r];
      float e2 = W_emb[2 * RED + r];
      dir[0][0] += a0 * e0; dir[0][1] += a0 * e1; dir[0][2] += a0 * e2;
      dir[1][0] += a1 * e0; dir[1][1] += a1 * e1; dir[1][2] += a1 * e2;
      dir[2][0] += a2 * e0; dir[2][1] += a2 * e1; dir[2][2] += a2 * e2;
    }
#pragma unroll
    for (int i = 0; i < 3; ++i) {
#pragma unroll
      for (int e = 0; e < 3; ++e) {
        float s = dir[i][e];
        s += __shfl_xor(s, 1);
        s += __shfl_xor(s, 2);
        dir[i][e] = s;
      }
    }
    float nd[3][3];
#pragma unroll
    for (int e = 0; e < 3; ++e) {
      float dn = sqrtf(dir[0][e] * dir[0][e] + dir[1][e] * dir[1][e] +
                       dir[2][e] * dir[2][e]) + EPS_N;
      float ri = 1.f / dn;
      nd[0][e] = dir[0][e] * ri;
      nd[1][e] = dir[1][e] * ri;
      nd[2][e] = dir[2][e] * ri;
    }
    for (int rr = 0; rr < 32; ++rr) {
      int r = q * 32 + rr;
      float a0 = vred[(3 * n + 0) * 132 + r];
      float a1 = vred[(3 * n + 1) * 132 + r];
      float a2 = vred[(3 * n + 2) * 132 + r];
      float nrm = sqrtf(a0 * a0 + a1 * a1 + a2 * a2);
      float inv = 1.f / (nrm + EPS_N);
      storeSplit(v2hi, v2lo, n * 520 + r, nrm);
#pragma unroll
      for (int j = 0; j < 3; ++j) {
        float p = (a0 * nd[0][j] + a1 * nd[1][j] + a2 * nd[2][j]) * inv;
        storeSplit(v2hi, v2lo, n * 520 + 128 + j * 128 + r, p);
      }
    }
  }
  __syncthreads();

  // ============== Phase B: GEMM2  (64x512)·(512x512), split-bf16 ===========
  f32x8 acc2[16];
#pragma unroll
  for (int t = 0; t < 16; ++t) acc2[t] = zero8;

  for (int kt = 0; kt < 16; ++kt) {
    if (kt < 15) {                         // prefetch next k-step's B tiles
#pragma unroll
      for (int ntl = 0; ntl < 4; ++ntl) {
        int nt = w * 4 + ntl;
        __builtin_prefetch(ws + OFF_B2HI + ((nt * 16 + kt + 1) * 32 + lane) * 8, 0, 3);
        __builtin_prefetch(ws + OFF_B2LO + ((nt * 16 + kt + 1) * 32 + lane) * 8, 0, 3);
      }
    }
    Frag ah[4], al[4];
    int kb = kt * 32 + khalf * 16;
#pragma unroll
    for (int mt = 0; mt < 4; ++mt) {
      int row = mt * 16 + m16;
      loadFragLDS(ah[mt], v2hi + row * 520 + kb);
      loadFragLDS(al[mt], v2lo + row * 520 + kb);
    }
#pragma unroll
    for (int ntl = 0; ntl < 4; ++ntl) {    // wave w owns n-tiles 4w..4w+3
      int nt = w * 4 + ntl;
      Frag bh, bl;
      loadFragG(bh, ws + OFF_B2HI + ((nt * 16 + kt) * 32 + lane) * 8);
      loadFragG(bl, ws + OFF_B2LO + ((nt * 16 + kt) * 32 + lane) * 8);
#pragma unroll
      for (int mt = 0; mt < 4; ++mt) {
        f32x8 c = acc2[mt * 4 + ntl];
        c = wmma_bf16(ah[mt], bh, c);
        c = wmma_bf16(ah[mt], bl, c);
        c = wmma_bf16(al[mt], bh, c);
        acc2[mt * 4 + ntl] = c;
      }
    }
  }

  // bias + dump h (64 x 512) f32 to LDS (row stride 516)
#pragma unroll
  for (int ntl = 0; ntl < 4; ++ntl) {
    int col = (w * 4 + ntl) * 16 + m16;
    float bv = b_lin[col];
#pragma unroll
    for (int mt = 0; mt < 4; ++mt) {
#pragma unroll
      for (int j = 0; j < 8; ++j) {
        int m = mt * 16 + j + 8 * khalf;
        hbuf[m * 516 + col] = acc2[mt * 4 + ntl][j] + bv;
      }
    }
  }
  __syncthreads();

  // ============== LayerNorm + exact GELU, store out ========================
  {
    int n = tid >> 2;
    int q = tid & 3;
    float s = 0.f, s2 = 0.f;
    for (int t = 0; t < 128; ++t) {
      float x = hbuf[n * 516 + q * 128 + t];
      s  += x;
      s2 += x * x;
    }
    s  += __shfl_xor(s, 1);  s  += __shfl_xor(s, 2);
    s2 += __shfl_xor(s2, 1); s2 += __shfl_xor(s2, 2);
    float mu   = s * (1.f / (float)OUT);
    float var  = s2 * (1.f / (float)OUT) - mu * mu;
    float rstd = rsqrtf(var + LN_EPS);
    float* op = out + ((size_t)blockIdx.x * 64 + n) * OUT + q * 128;
    for (int t = 0; t < 128; t += 4) {
      float4 g  = *(const float4*)(gamma + q * 128 + t);
      float4 bb = *(const float4*)(beta  + q * 128 + t);
      float4 r;
      float x0 = (hbuf[n * 516 + q * 128 + t + 0] - mu) * rstd * g.x + bb.x;
      float x1 = (hbuf[n * 516 + q * 128 + t + 1] - mu) * rstd * g.y + bb.y;
      float x2 = (hbuf[n * 516 + q * 128 + t + 2] - mu) * rstd * g.z + bb.z;
      float x3 = (hbuf[n * 516 + q * 128 + t + 3] - mu) * rstd * g.w + bb.w;
      r.x = 0.5f * x0 * (1.f + erff(x0 * 0.70710678118654752f));
      r.y = 0.5f * x1 * (1.f + erff(x1 * 0.70710678118654752f));
      r.z = 0.5f * x2 * (1.f + erff(x2 * 0.70710678118654752f));
      r.w = 0.5f * x3 * (1.f + erff(x3 * 0.70710678118654752f));
      *(float4*)(op + t) = r;
    }
  }
}

// ---------------------------------------------------------------------------
// Launch
// ---------------------------------------------------------------------------
extern "C" void kernel_launch(void* const* d_in, const int* in_sizes, int n_in,
                              void* d_out, int out_size, void* d_ws, size_t ws_size,
                              hipStream_t stream) {
  const float* v   = (const float*)d_in[0];
  const float* Wd  = (const float*)d_in[1];
  const float* We  = (const float*)d_in[2];
  const float* Wl  = (const float*)d_in[3];
  const float* bl  = (const float*)d_in[4];
  const float* g   = (const float*)d_in[5];
  const float* be  = (const float*)d_in[6];
  uint32_t* ws = (uint32_t*)d_ws;
  float* out = (float*)d_out;

  (void)hipFuncSetAttribute(reinterpret_cast<const void*>(scalarizer_fused),
                            hipFuncAttributeMaxDynamicSharedMemorySize,
                            SMEM_BYTES);

  prep_weights<<<640, 256, 0, stream>>>(Wd, Wl, ws);
  scalarizer_fused<<<NROWS / 64, 256, SMEM_BYTES, stream>>>(v, We, bl, g, be, ws, out);
}
